// MambaSSM_79465484910713
// MI455X (gfx1250) — compile-verified, matrix operand order checked by reading, not run on hardware
//
#include <hip/hip_runtime.h>

// ---------------- problem constants ----------------
constexpr int BB   = 2;      // batch
constexpr int LL   = 2048;   // sequence length
constexpr int DM   = 1024;   // d_model
constexpr int CC   = 2048;   // expanded channels (EXPAND=2)
constexpr int PP   = 96;     // DT_RANK + 2*D_STATE
constexpr int RR   = 64;     // DT_RANK
constexpr int NST  = 16;     // D_STATE
constexpr int NC   = 32;     // scan chunks
constexpr int LC   = 64;     // chunk length (NC*LC == LL)

// ---------------- WMMA types ----------------
typedef __attribute__((ext_vector_type(16))) _Float16 v16h;
typedef __attribute__((ext_vector_type(8)))  _Float16 v8h;
typedef __attribute__((ext_vector_type(2)))  _Float16 v2h;
typedef __attribute__((ext_vector_type(8)))  float    v8f;

// A-fragment loader for V_WMMA_F32_16X16X32_F16 (ISA 7.12.2, 16-bit A 16x32):
// lane l<16 : row M=l,   K = {kbase+0..7}  and {kbase+16..23}
// lane l>=16: row M=l-16,K = {kbase+8..15} and {kbase+24..31}
// => two contiguous 16-byte loads per lane. B-fragment uses the same layout
// applied to B^T (row-major NxK weight matrix), by lane<->N symmetry with C.
__device__ __forceinline__ v16h load_frag16(const _Float16* __restrict__ base,
                                            int rowStride, int lane, int kbase) {
  int row  = lane & 15;
  int koff = (lane >> 4) << 3;           // 0 or 8
  const _Float16* p = base + (long)row * rowStride + kbase + koff;
  union { v16h w; v8h h[2]; } u;
  u.h[0] = *(const v8h*)(p);
  u.h[1] = *(const v8h*)(p + 16);
  return u.w;
}

// ---------------- kernel 0: weight conversion to f16 ----------------
__global__ void mamba_cvt_weights(const float* __restrict__ wx,   // (96,2048)
                                  const float* __restrict__ wdt,  // (1024,64)
                                  _Float16* __restrict__ wx16,
                                  _Float16* __restrict__ wdt16) {
  int i = blockIdx.x * blockDim.x + threadIdx.x;
  const int NW1 = PP * CC;
  const int NW2 = DM * RR;
  if (i < NW1)            wx16[i]        = (_Float16)wx[i];
  else if (i < NW1 + NW2) wdt16[i - NW1] = (_Float16)wdt[i - NW1];
}

// ---------------- kernel 1: causal depthwise conv + SiLU ----------------
// writes x_conv as f16 (GEMM A operand) and x_mean as f32, for both directions
__global__ void mamba_conv_silu(const float* __restrict__ x,      // (B,L,DM)
                                const float* __restrict__ cw,     // (CC,1,4)
                                const float* __restrict__ cb,     // (CC)
                                _Float16* __restrict__ aconv,     // (2,B,L,CC) f16
                                float* __restrict__ xmean) {      // (2,B,L,DM)
  long idx = (long)blockIdx.x * blockDim.x + threadIdx.x;
  const long N = 2L * BB * LL * DM;
  if (idx >= N) return;
  int d = (int)(idx % DM); long r = idx / DM;
  int t = (int)(r % LL);   r /= LL;
  int b = (int)(r % BB);   int dir = (int)(r / BB);

  float xs[4];
#pragma unroll
  for (int k = 0; k < 4; ++k) {
    int ts = t - 3 + k;                 // causal tap
    float v = 0.f;
    if (ts >= 0) {
      int torig = dir ? (LL - 1 - ts) : ts;
      v = x[((long)b * LL + torig) * DM + d];
    }
    xs[k] = v;
  }
  int c0 = 2 * d;
  float a0 = cb[c0], a1 = cb[c0 + 1];
#pragma unroll
  for (int k = 0; k < 4; ++k) {
    a0 += cw[c0 * 4 + k]       * xs[k];
    a1 += cw[(c0 + 1) * 4 + k] * xs[k];
  }
  float s0 = a0 / (1.f + __expf(-a0));  // SiLU
  float s1 = a1 / (1.f + __expf(-a1));
  long o = (((long)dir * BB + b) * LL + t) * CC + c0;
  v2h h2; h2[0] = (_Float16)s0; h2[1] = (_Float16)s1;
  *(v2h*)(aconv + o) = h2;
  xmean[(((long)dir * BB + b) * LL + t) * DM + d] = 0.5f * (s0 + s1);
}

// ---------------- kernel 2: GEMM1 x_dbl = x_conv * Wx^T (WMMA f16->f32) ----
__global__ __launch_bounds__(256)
void mamba_gemm_xproj(const _Float16* __restrict__ A,     // (2,B,L,CC) f16
                      const _Float16* __restrict__ Wx,    // (96,CC) f16
                      float* __restrict__ xdbl,           // (2,B,L,96)
                      _Float16* __restrict__ dtpart) {    // (2,B,L,64) f16
  int wave = (blockIdx.x * blockDim.x + threadIdx.x) >> 5;
  int lane = threadIdx.x & 31;
  const int NT = PP / 16;        // 6
  const int MT = LL / 16;        // 128
  if (wave >= 2 * BB * MT * NT) return;
  int nt = wave % NT; int r = wave / NT;
  int mt = r % MT;    r /= MT;
  int b  = r % BB;    int dir = r / BB;

  const _Float16* abase = A  + (((long)dir * BB + b) * LL + (long)mt * 16) * CC;
  const _Float16* bbase = Wx + (long)nt * 16 * CC;

  v8f acc = {};
  for (int k = 0; k < CC; k += 32) {
    if (k + 64 < CC)
      __builtin_prefetch(abase + (long)(lane & 15) * CC + k + 64, 0, 1);
    v16h af = load_frag16(abase, CC, lane, k);
    v16h bf = load_frag16(bbase, CC, lane, k);
    acc = __builtin_amdgcn_wmma_f32_16x16x32_f16(false, af, false, bf,
                                                 (short)0, acc, false, false);
  }
  // C/D layout: VGPR r -> M = r + 8*(lane>=16), N = lane&15
  int n = lane & 15;
  int mh = (lane >> 4) * 8;
  long trow = ((long)dir * BB + b) * LL + (long)mt * 16;
#pragma unroll
  for (int rI = 0; rI < 8; ++rI) {
    int  m = rI + mh;
    long t = trow + m;
    float v = acc[rI];
    xdbl[t * PP + nt * 16 + n] = v;
    if (nt < 4) dtpart[t * RR + nt * 16 + n] = (_Float16)v;  // dt_part copy (f16)
  }
}

// ---------------- kernel 3: GEMM2 dt = softplus(dt_part*Wdt^T + b) (WMMA) ---
__global__ __launch_bounds__(256)
void mamba_gemm_dt(const _Float16* __restrict__ Ain,      // (2,B,L,64) f16
                   const _Float16* __restrict__ Wdt,      // (1024,64) f16
                   const float* __restrict__ bias,        // (1024)
                   float* __restrict__ dt) {              // (2,B,L,1024)
  int wave = (blockIdx.x * blockDim.x + threadIdx.x) >> 5;
  int lane = threadIdx.x & 31;
  const int NT = DM / 16;        // 64
  const int MT = LL / 16;        // 128
  if (wave >= 2 * BB * MT * NT) return;
  int nt = wave % NT; int r = wave / NT;
  int mt = r % MT;    r /= MT;
  int b  = r % BB;    int dir = r / BB;

  const _Float16* abase = Ain + (((long)dir * BB + b) * LL + (long)mt * 16) * RR;
  const _Float16* bbase = Wdt + (long)nt * 16 * RR;

  v8f acc = {};
#pragma unroll
  for (int k = 0; k < RR; k += 32) {
    v16h af = load_frag16(abase, RR, lane, k);
    v16h bf = load_frag16(bbase, RR, lane, k);
    acc = __builtin_amdgcn_wmma_f32_16x16x32_f16(false, af, false, bf,
                                                 (short)0, acc, false, false);
  }
  int n = lane & 15;
  int nglob = nt * 16 + n;
  int mh = (lane >> 4) * 8;
  float bn = bias[nglob];
  long trow = ((long)dir * BB + b) * LL + (long)mt * 16;
#pragma unroll
  for (int rI = 0; rI < 8; ++rI) {
    long t = trow + rI + mh;
    float z = acc[rI] + bn;
    float sp = (z > 20.f) ? z : log1pf(__expf(z));        // softplus
    dt[t * DM + nglob] = sp;
  }
}

// ---------------- kernel 4: scan phase 1 (per-chunk, zero init) -------------
// per (dir,b,chunk,d): local scan with h=0; emit decay-product + local end-state
__global__ void mamba_scan_chunk(const float* __restrict__ dt,     // (2,B,L,DM)
                                 const float* __restrict__ xdbl,   // (2,B,L,96)
                                 const float* __restrict__ xmean,  // (2,B,L,DM)
                                 const float* __restrict__ A_log,  // (DM,16)
                                 float* __restrict__ aprod,        // (2,B,NC,DM,16)
                                 float* __restrict__ hend) {       // same
  long idx = (long)blockIdx.x * blockDim.x + threadIdx.x;
  const long N = 2L * BB * NC * DM;
  if (idx >= N) return;
  int d  = (int)(idx % DM); long r = idx / DM;
  int ch = (int)(r % NC);   r /= NC;
  int b  = (int)(r % BB);   int dir = (int)(r / BB);

  float Av[NST], h[NST], ap[NST];
#pragma unroll
  for (int n = 0; n < NST; ++n) {
    Av[n] = -__expf(A_log[d * NST + n]);
    h[n] = 0.f; ap[n] = 1.f;
  }
  long seq = (long)dir * BB + b;
  for (int tl = 0; tl < LC; ++tl) {
    long ro  = seq * LL + (long)ch * LC + tl;
    float dtv = dt[ro * DM + d];
    float xm  = xmean[ro * DM + d];
    const float4* Bp = (const float4*)(xdbl + ro * PP + RR);      // B_mod (16)
    float4 b4[4] = { Bp[0], Bp[1], Bp[2], Bp[3] };
    const float* Bs = (const float*)b4;
    float u = dtv * xm;
#pragma unroll
    for (int n = 0; n < NST; ++n) {
      float dec = __expf(dtv * Av[n]);
      h[n]  = h[n] * dec + u * Bs[n];
      ap[n] *= dec;
    }
  }
  long o = ((seq * NC + ch) * DM + d) * NST;
#pragma unroll
  for (int j = 0; j < 4; ++j) {
    ((float4*)(aprod + o))[j] = make_float4(ap[4*j], ap[4*j+1], ap[4*j+2], ap[4*j+3]);
    ((float4*)(hend  + o))[j] = make_float4(h[4*j],  h[4*j+1],  h[4*j+2],  h[4*j+3]);
  }
}

// ---------------- kernel 5: scan phase 2 (chunk-level recurrence) -----------
__global__ void mamba_scan_combine(const float* __restrict__ aprod,
                                   const float* __restrict__ hend,
                                   float* __restrict__ hin) {      // (2,B,NC,DM,16)
  long idx = (long)blockIdx.x * blockDim.x + threadIdx.x;
  const long N = 2L * BB * DM;
  if (idx >= N) return;
  int d = (int)(idx % DM); long r = idx / DM;
  int b = (int)(r % BB);   int dir = (int)(r / BB);

  float h[NST];
#pragma unroll
  for (int n = 0; n < NST; ++n) h[n] = 0.f;
  long seq = (long)dir * BB + b;
  for (int c = 0; c < NC; ++c) {
    long o = ((seq * NC + c) * DM + d) * NST;
#pragma unroll
    for (int j = 0; j < 4; ++j)
      ((float4*)(hin + o))[j] = make_float4(h[4*j], h[4*j+1], h[4*j+2], h[4*j+3]);
#pragma unroll
    for (int n = 0; n < NST; ++n)
      h[n] = aprod[o + n] * h[n] + hend[o + n];
  }
}

// ---------------- kernel 6: scan phase 3 (replay with true h_in, emit y) ----
__global__ void mamba_scan_final(const float* __restrict__ dt,
                                 const float* __restrict__ xdbl,
                                 const float* __restrict__ xmean,
                                 const float* __restrict__ A_log,
                                 const float* __restrict__ hin,
                                 const float* __restrict__ Dp,
                                 float* __restrict__ out,          // (B,L,DM)
                                 int dir, int accumulate) {
  long idx = (long)blockIdx.x * blockDim.x + threadIdx.x;
  const long N = (long)BB * NC * DM;
  if (idx >= N) return;
  int d  = (int)(idx % DM); long r = idx / DM;
  int ch = (int)(r % NC);
  int b  = (int)(r / NC);

  long seq = (long)dir * BB + b;
  long o = ((seq * NC + ch) * DM + d) * NST;
  float Av[NST], h[NST];
#pragma unroll
  for (int n = 0; n < NST; ++n) {
    Av[n] = -__expf(A_log[d * NST + n]);
    h[n] = hin[o + n];
  }
  float Dd = Dp[d];
  for (int tl = 0; tl < LC; ++tl) {
    int  t  = ch * LC + tl;
    long ro = seq * LL + t;
    float dtv = dt[ro * DM + d];
    float xm  = xmean[ro * DM + d];
    const float4* Bp = (const float4*)(xdbl + ro * PP + RR);       // B_mod
    const float4* Cp = (const float4*)(xdbl + ro * PP + RR + NST); // C_mod
    float4 b4[4] = { Bp[0], Bp[1], Bp[2], Bp[3] };
    float4 c4[4] = { Cp[0], Cp[1], Cp[2], Cp[3] };
    const float* Bs = (const float*)b4;
    const float* Cs = (const float*)c4;
    float u = dtv * xm;
    float y = 0.f;
#pragma unroll
    for (int n = 0; n < NST; ++n) {
      float dec = __expf(dtv * Av[n]);
      h[n] = h[n] * dec + u * Bs[n];
      y   += h[n] * Cs[n];
    }
    y += Dd * xm;
    int torig = dir ? (LL - 1 - t) : t;
    long oo = ((long)b * LL + torig) * DM + d;
    if (accumulate) out[oo] += y; else out[oo] = y;
  }
}

// ---------------- host launch ----------------
extern "C" void kernel_launch(void* const* d_in, const int* in_sizes, int n_in,
                              void* d_out, int out_size, void* d_ws, size_t ws_size,
                              hipStream_t stream) {
  (void)in_sizes; (void)n_in; (void)out_size; (void)ws_size;
  const float* x       = (const float*)d_in[0];
  const float* conv_w  = (const float*)d_in[1];
  const float* conv_b  = (const float*)d_in[2];
  const float* xproj_w = (const float*)d_in[3];
  const float* dtproj_w= (const float*)d_in[4];
  const float* dtproj_b= (const float*)d_in[5];
  const float* A_log   = (const float*)d_in[6];
  const float* Dp      = (const float*)d_in[7];
  float* out = (float*)d_out;

  char* ws = (char*)d_ws;
  size_t off = 0;
  auto alloc = [&](size_t bytes) {
    off = (off + 255) & ~(size_t)255;
    size_t o = off; off += bytes; return o;
  };
  _Float16* wx16   = (_Float16*)(ws + alloc((size_t)PP * CC * 2));
  _Float16* wdt16  = (_Float16*)(ws + alloc((size_t)DM * RR * 2));
  _Float16* aconv  = (_Float16*)(ws + alloc(2ULL * BB * LL * CC * 2));
  float*    xmean  = (float*)   (ws + alloc(2ULL * BB * LL * DM * 4));
  float*    xdbl   = (float*)   (ws + alloc(2ULL * BB * LL * PP * 4));
  _Float16* dtpart = (_Float16*)(ws + alloc(2ULL * BB * LL * RR * 2));
  float*    dtbuf  = (float*)   (ws + alloc(2ULL * BB * LL * DM * 4));
  float*    aprod  = (float*)   (ws + alloc(2ULL * BB * NC * DM * NST * 4));
  float*    hendb  = (float*)   (ws + alloc(2ULL * BB * NC * DM * NST * 4));
  float*    hinb   = (float*)   (ws + alloc(2ULL * BB * NC * DM * NST * 4));

  // 0) weights -> f16
  {
    int n = PP * CC + DM * RR;
    mamba_cvt_weights<<<(n + 255) / 256, 256, 0, stream>>>(xproj_w, dtproj_w, wx16, wdt16);
  }
  // 1) conv + SiLU (both directions)
  {
    long n = 2L * BB * LL * DM;
    mamba_conv_silu<<<(int)((n + 255) / 256), 256, 0, stream>>>(x, conv_w, conv_b, aconv, xmean);
  }
  // 2) GEMM1 (WMMA): x_dbl
  {
    int waves = 2 * BB * (LL / 16) * (PP / 16);     // 3072
    mamba_gemm_xproj<<<(waves * 32 + 255) / 256, 256, 0, stream>>>(aconv, wx16, xdbl, dtpart);
  }
  // 3) GEMM2 (WMMA): dt = softplus(...)
  {
    int waves = 2 * BB * (LL / 16) * (DM / 16);     // 32768
    mamba_gemm_dt<<<(waves * 32 + 255) / 256, 256, 0, stream>>>(dtpart, wdt16, dtproj_b, dtbuf);
  }
  // 4) chunked scan, phase 1
  {
    long n = 2L * BB * NC * DM;
    mamba_scan_chunk<<<(int)((n + 255) / 256), 256, 0, stream>>>(dtbuf, xdbl, xmean, A_log, aprod, hendb);
  }
  // 5) chunk-level combine
  {
    long n = 2L * BB * DM;
    mamba_scan_combine<<<(int)((n + 255) / 256), 256, 0, stream>>>(aprod, hendb, hinb);
  }
  // 6) replay with true incoming state; dir0 writes, dir1 accumulates
  {
    long n = (long)BB * NC * DM;
    mamba_scan_final<<<(int)((n + 255) / 256), 256, 0, stream>>>(dtbuf, xdbl, xmean, A_log, hinb, Dp, out, 0, 0);
    mamba_scan_final<<<(int)((n + 255) / 256), 256, 0, stream>>>(dtbuf, xdbl, xmean, A_log, hinb, Dp, out, 1, 1);
  }
}
